// GameboyNet_74612171866120
// MI455X (gfx1250) — matrix operand order repaired and checked by reading, not compiled, not run on hardware
//
#include <hip/hip_runtime.h>
#include <cmath>

// ---------------------------------------------------------------------------
// Problem constants (match reference)
// ---------------------------------------------------------------------------
#define BB   8
#define SS   4096
#define DD   256
#define WW   512
#define LL   32
#define EE   1024
#define MM   (BB * SS)     // 32768 token rows
#define NWW  (SS / WW)     // 8 windows

typedef __attribute__((ext_vector_type(16))) _Float16 v16h;
typedef __attribute__((ext_vector_type(8)))  _Float16 v8h;
typedef __attribute__((ext_vector_type(4)))  _Float16 v4h;
typedef __attribute__((ext_vector_type(8)))  float    v8f;
typedef __attribute__((ext_vector_type(4)))  unsigned int u32x4;
typedef __attribute__((ext_vector_type(8)))  int          i32x8;
typedef __attribute__((ext_vector_type(4)))  int          i32x4;

#ifndef __has_builtin
#define __has_builtin(x) 0
#endif
#if __has_builtin(__builtin_amdgcn_tensor_load_to_lds) && \
    __has_builtin(__builtin_amdgcn_s_wait_tensorcnt)
#define USE_TDM 1
#else
#define USE_TDM 0
#endif

// ---------------------------------------------------------------------------
// WMMA fragment helpers (CDNA5 16x16x32 f16 layouts, wave32)
//   A frag 16x32 : lane = M-row (l&15); halves K=kb..kb+7 and kb+16..kb+23,
//                  kb = (l<16)?0:8  -> two contiguous 16B LDS reads
//   B frag 32x16 : lane = N-col (l&15); halves K=kb..kb+7 and kb+16..kb+23, kb=(l<16)?0:8
//                  -> two contiguous 16B LDS reads (needs N-major, K-contig)
//   C/D 16x16    : vgpr e, lane l -> M = e + (l>>4)*8, N = l&15
// ---------------------------------------------------------------------------
__device__ __forceinline__ v16h frag_cat(v8h lo, v8h hi) {
  return __builtin_shufflevector(lo, hi, 0,1,2,3,4,5,6,7,8,9,10,11,12,13,14,15);
}

__device__ __forceinline__ v16h load_a_frag(const _Float16* tile, int ld,
                                            int row0, int k0, int lane) {
  int r  = row0 + (lane & 15);
  int kb = k0 + ((lane & 16) ? 8 : 0);
  const _Float16* p = tile + r * ld + kb;
  return frag_cat(*(const v8h*)p, *(const v8h*)(p + 16));
}

__device__ __forceinline__ v16h load_b_frag(const _Float16* tile, int ld,
                                            int n0, int k0, int lane) {
  int n  = n0 + (lane & 15);
  int kb = k0 + ((lane & 16) ? 8 : 0);
  const _Float16* p = tile + n * ld + kb;
  return frag_cat(*(const v8h*)p, *(const v8h*)(p + 16));
}

__device__ __forceinline__ v8f wmma_f16(v16h a, v16h b, v8f c) {
  return __builtin_amdgcn_wmma_f32_16x16x32_f16(false, a, false, b,
                                                (short)0, c, false, false);
}

#if USE_TDM
// ---------------------------------------------------------------------------
// Tensor Data Mover: 2D tile (f16 elements) global -> LDS.
//   tile_w elements per row, tile_h rows, row stride = stride_elems.
//   Issued per-wave (EXEC-ignored); tracked by TENSORcnt.
//   D# built per CDNA5 ISA (group0 + group1, 2D mode).
// ---------------------------------------------------------------------------
__device__ __forceinline__ void tdm_load_f16(unsigned int lds_off,
                                             const void* gptr,
                                             unsigned int tile_w,
                                             unsigned int tile_h,
                                             unsigned int stride_elems) {
  unsigned long long ga = (unsigned long long)(uintptr_t)gptr;
  u32x4 g0;
  g0.x = 1u;                                   // count=1 (valid user descriptor)
  g0.y = lds_off;                              // lds_addr
  g0.z = (unsigned int)ga;                     // global_addr[31:0]
  g0.w = (unsigned int)((ga >> 32) & 0x01FFFFFFu) | (2u << 30);  // [56:32]|type=2
  i32x8 g1;
  g1[0] = (int)(1u << 16);                     // data_size=1 (2 bytes), no flags
  g1[1] = (int)0xFFFF0000u;                    // tensor_dim0 = 0x7FFFFFFF (lo16)
  g1[2] = (int)0xFFFF7FFFu;                    // tensor_dim0 hi | tensor_dim1 lo
  g1[3] = (int)(0x7FFFu | (tile_w << 16));     // tensor_dim1 hi | tile_dim0
  g1[4] = (int)(tile_h & 0xFFFFu);             // tile_dim1 | tile_dim2=0
  g1[5] = (int)stride_elems;                   // tensor_dim0_stride[31:0]
  g1[6] = 0;                                   // stride0 hi | stride1 lo
  g1[7] = 0;                                   // stride1 hi
  i32x4 gz = {0, 0, 0, 0};
#if defined(__clang_major__) && (__clang_major__ >= 23)
  i32x8 gz8 = {0, 0, 0, 0, 0, 0, 0, 0};
  __builtin_amdgcn_tensor_load_to_lds(g0, g1, gz, gz, gz8, 0);
#else
  __builtin_amdgcn_tensor_load_to_lds(g0, g1, gz, gz, 0);
#endif
}
#endif  // USE_TDM

// ---------------------------------------------------------------------------
// Embedding gather: h = emb[x]  (fp32 + f16 shadow copy)
// ---------------------------------------------------------------------------
__global__ __launch_bounds__(DD) void embed_kernel(
    const int* __restrict__ x, const float* __restrict__ emb,
    float* __restrict__ h32, _Float16* __restrict__ h16) {
  int tok = blockIdx.x;
  int d   = threadIdx.x;
  int id  = x[tok];
  float v = emb[id * DD + d];
  h32[(size_t)tok * DD + d] = v;
  h16[(size_t)tok * DD + d] = (_Float16)v;
}

// ---------------------------------------------------------------------------
// Weight conversion fp32 -> f16 (once per layer, weights live in L2)
// ---------------------------------------------------------------------------
__global__ __launch_bounds__(256) void convw_kernel(
    const float* __restrict__ src, _Float16* __restrict__ dst) {
  const int i = blockIdx.x * 256 + threadIdx.x;   // one float4 per thread
  float4 f = ((const float4*)src)[i];
  v4h h;
  h[0] = (_Float16)f.x; h[1] = (_Float16)f.y;
  h[2] = (_Float16)f.z; h[3] = (_Float16)f.w;
  ((v4h*)dst)[i] = h;
}

// ---------------------------------------------------------------------------
// GEMM: C[M,N] = epilogue(A_f16[M,K] @ B_f16[N,K]^T + bias[N])
//   block tile 128x64, 8 waves in a 4(M)x2(N) grid, 32x32 per wave
//   K tiles double-buffered via TDM (tensor_load_to_lds + s_wait_tensorcnt)
//   EPI 0: ->f16              (QKV projections)
//   EPI 1: sigmoid ->f16      (FFN up)
//   EPI 2: +res, BN ->f32+f16 (FFN down)
//   EPI 3: relu, transpose to (B,D,S) ->f32  (final projection)
// ---------------------------------------------------------------------------
template <int EPI>
__global__ __launch_bounds__(256) void gemm_kernel(
    const _Float16* __restrict__ A, const _Float16* __restrict__ Bw,
    const float* __restrict__ bias, int K, int N,
    _Float16* __restrict__ out16, float* __restrict__ out32,
    const float* __restrict__ gamma, const float* __restrict__ beta,
    float bn_scale) {
  const int tid = threadIdx.x, lane = tid & 31, wave = tid >> 5;
  const int wm = wave & 3, wn = wave >> 2;
  const int m0 = blockIdx.x * 128, n0 = blockIdx.y * 64;
  const int nIter = K >> 5;

  v8f acc00 = {}, acc01 = {}, acc10 = {}, acc11 = {};

  auto mm_step = [&](const _Float16* Ab, const _Float16* Bb) {
    v16h af0 = load_a_frag(Ab, 32, wm * 32 + 0,  0, lane);
    v16h af1 = load_a_frag(Ab, 32, wm * 32 + 16, 0, lane);
    v16h bf0 = load_b_frag(Bb, 32, wn * 32 + 0,  0, lane);
    v16h bf1 = load_b_frag(Bb, 32, wn * 32 + 16, 0, lane);
    acc00 = wmma_f16(af0, bf0, acc00);
    acc01 = wmma_f16(af0, bf1, acc01);
    acc10 = wmma_f16(af1, bf0, acc10);
    acc11 = wmma_f16(af1, bf1, acc11);
  };

#if USE_TDM
  __shared__ _Float16 As[2][128 * 32];  // 2 x 8 KB
  __shared__ _Float16 Bs[2][64 * 32];   // 2 x 4 KB
  auto issue_tile = [&](int it) {       // wave0 only: two DMA descriptors
    const int k0 = it << 5, buf = it & 1;
    tdm_load_f16((unsigned int)(uintptr_t)&As[buf][0],
                 A + (size_t)m0 * K + k0, 32u, 128u, (unsigned int)K);
    tdm_load_f16((unsigned int)(uintptr_t)&Bs[buf][0],
                 Bw + (size_t)n0 * K + k0, 32u, 64u, (unsigned int)K);
  };
  if (wave == 0) issue_tile(0);
  for (int it = 0; it < nIter; ++it) {
    if (wave == 0) {
      if (it + 1 < nIter) {
        issue_tile(it + 1);                       // prefetch next tile
        __builtin_amdgcn_s_wait_tensorcnt(2);     // tile `it` complete (in-order)
      } else {
        __builtin_amdgcn_s_wait_tensorcnt(0);
      }
    }
    __syncthreads();
    mm_step(&As[it & 1][0], &Bs[it & 1][0]);
    __syncthreads();                              // tile consumed -> reusable
  }
#else
  __shared__ _Float16 As1[128 * 32];
  __shared__ _Float16 Bs1[64 * 32];
  const int ar = tid >> 1, ac = (tid & 1) * 16;   // 16 halves/thread
  const int br = tid >> 2, bc = (tid & 3) * 8;    // 8 halves/thread
  for (int it = 0; it < nIter; ++it) {
    const int k0 = it << 5;
    const uint4* ga = (const uint4*)(A + (size_t)(m0 + ar) * K + k0 + ac);
    uint4 a0 = ga[0], a1 = ga[1];
    *(uint4*)&As1[ar * 32 + ac]     = a0;
    *(uint4*)&As1[ar * 32 + ac + 8] = a1;
    *(uint4*)&Bs1[br * 32 + bc] =
        *(const uint4*)(Bw + (size_t)(n0 + br) * K + k0 + bc);
    __syncthreads();
    mm_step(As1, Bs1);
    __syncthreads();
  }
#endif

  // -- epilogue
  const int hi = lane >> 4, nn = lane & 15;
  v8f accs[2][2] = {{acc00, acc01}, {acc10, acc11}};
  for (int i = 0; i < 2; ++i) {
    for (int j = 0; j < 2; ++j) {
      v8f acc = accs[i][j];
      const int mbase = m0 + wm * 32 + i * 16 + hi * 8;
      const int nidx  = n0 + wn * 32 + j * 16 + nn;
      const float bval = bias[nidx];
      for (int e = 0; e < 8; ++e) {
        const int m = mbase + e;
        float v = acc[e] + bval;
        const size_t idx = (size_t)m * N + nidx;
        if constexpr (EPI == 0) {
          out16[idx] = (_Float16)v;
        } else if constexpr (EPI == 1) {
          out16[idx] = (_Float16)(1.f / (1.f + __expf(-v)));
        } else if constexpr (EPI == 2) {
          float hv = out32[idx] + v;                       // residual
          hv = gamma[nidx] * (hv * bn_scale) + beta[nidx]; // BN (eval)
          out32[idx] = hv;
          out16[idx] = (_Float16)hv;
        } else {  // EPI == 3: relu + transpose (B,S,D) -> (B,D,S)
          const int bb = m >> 12;        // m / S
          const int ss = m & (SS - 1);   // m % S
          out32[((size_t)bb * DD + nidx) * SS + ss] = v > 0.f ? v : 0.f;
        }
      }
    }
  }
}

// ---------------------------------------------------------------------------
// Windowed causal attention (flash-style), residual-added into h.
//   grid = (W/32, NW, B); block = 64 threads (2 waves x 16 query rows)
//   keys j in [0,1024) map to global (nw-1)*W + j; mask: j <= W + qo + r
//   Q and K tiles DMA'd by TDM; V transposed manually (overlaps K DMA).
// ---------------------------------------------------------------------------
__global__ __launch_bounds__(64) void attn_kernel(
    const _Float16* __restrict__ q16, const _Float16* __restrict__ k16,
    const _Float16* __restrict__ v16p, float* __restrict__ h32,
    _Float16* __restrict__ h16) {
  __shared__ _Float16 Qs[32 * 256];   // 16 KB, row-major q x d
  __shared__ _Float16 Ks[32 * 256];   // 16 KB, row-major k x d
  __shared__ _Float16 Vt[256 * 32];   // 16 KB, transposed d x k
  __shared__ float    Ss[2][16 * 32]; //  4 KB, per-wave raw scores
  __shared__ _Float16 Ps[2][16 * 32]; //  2 KB, per-wave exp probs (f16)
  __shared__ float row_m[32], row_l[32], row_a[32];

  const int tid = threadIdx.x, lane = tid & 31, wave = tid >> 5;
  const int qo = blockIdx.x * 32;          // query offset within window
  const int nw = blockIdx.y, bidx = blockIdx.z;
  const size_t qrow0 = (size_t)bidx * SS + nw * WW + qo;
  const int r0 = wave * 16;
  const int hi = lane >> 4, nn = lane & 15;

  // stage Q (32 rows x 256 halves, rows contiguous -> flat 16 KB tile)
#if USE_TDM
  if (wave == 0)
    tdm_load_f16((unsigned int)(uintptr_t)&Qs[0], q16 + qrow0 * DD,
                 8192u, 1u, 8192u);
#else
  for (int i = tid; i < 32 * 16; i += 64) {
    const int r = i >> 4, c = (i & 15) * 16;
    *(uint4*)&Qs[r * 256 + c]     = *(const uint4*)&q16[(qrow0 + r) * DD + c];
    *(uint4*)&Qs[r * 256 + c + 8] = *(const uint4*)&q16[(qrow0 + r) * DD + c + 8];
  }
#endif
  if (tid < 32) { row_m[tid] = -1e30f; row_l[tid] = 0.f; }

  v8f O[16];
  {
    v8f z = {};
    for (int i = 0; i < 16; ++i) O[i] = z;
  }

  const int t_start = (nw == 0) ? 16 : 0;  // window 0: prev window fully masked
  int t_end = (WW + qo + 31) / 32 + 1;
  if (t_end > 32) t_end = 32;
  const int kbase_g = (nw - 1) * WW;

  for (int t = t_start; t < t_end; ++t) {
    const int j0 = t * 32;
    __syncthreads();  // previous K/V tile fully consumed

    const size_t krow0 = (size_t)bidx * SS + kbase_g + j0;
#if USE_TDM
    if (wave == 0)    // K tile is 32 contiguous rows -> flat 16 KB DMA
      tdm_load_f16((unsigned int)(uintptr_t)&Ks[0], k16 + krow0 * DD,
                   8192u, 1u, 8192u);
#else
    for (int i = tid; i < 32 * 16; i += 64) {
      const int r = i >> 4, c = (i & 15) * 16;
      *(uint4*)&Ks[r * 256 + c]     = *(const uint4*)&k16[(krow0 + r) * DD + c];
      *(uint4*)&Ks[r * 256 + c + 8] = *(const uint4*)&k16[(krow0 + r) * DD + c + 8];
    }
#endif
    // V transpose (manual; overlaps the K DMA)
    for (int i = tid; i < 32 * 32; i += 64) {
      const int j = i >> 5, dc = (i & 31) * 8;
      v8h vv = *(const v8h*)&v16p[(krow0 + j) * DD + dc];
      for (int u = 0; u < 8; ++u) Vt[(dc + u) * 32 + j] = vv[u];
    }
#if USE_TDM
    if (wave == 0) __builtin_amdgcn_s_wait_tensorcnt(0);  // Q (1st iter) + K
#endif
    __syncthreads();

    // S = Q K^T over D=256 (8 chunks of 32)
    v8f s0 = {}, s1 = {};
    for (int kc = 0; kc < 8; ++kc) {
      v16h a  = load_a_frag(Qs, 256, r0, kc * 32, lane);
      v16h b0 = load_b_frag(Ks, 256, 0,  kc * 32, lane);
      v16h b1 = load_b_frag(Ks, 256, 16, kc * 32, lane);
      s0 = wmma_f16(a, b0, s0);
      s1 = wmma_f16(a, b1, s1);
    }
    // scale + causal-window mask, spill to LDS for row-wise softmax
    for (int e = 0; e < 8; ++e) {
      const int rloc = e + hi * 8;
      const int lim  = WW + qo + r0 + rloc;  // key index limit (inclusive)
      const int ja = j0 + nn, jb = j0 + 16 + nn;
      Ss[wave][rloc * 32 + nn]      = (ja <= lim) ? s0[e] * 0.0625f : -1e30f;
      Ss[wave][rloc * 32 + 16 + nn] = (jb <= lim) ? s1[e] * 0.0625f : -1e30f;
    }
    __syncthreads();

    // online softmax state update: lanes 0..15 own one row each
    if (lane < 16) {
      const int rblk = r0 + lane;
      const float m_old = row_m[rblk];
      float mx = m_old;
      const float* srow = &Ss[wave][lane * 32];
      for (int j = 0; j < 32; ++j) mx = fmaxf(mx, srow[j]);
      const float alpha = __expf(m_old - mx);
      float sum = 0.f;
      _Float16* prow = &Ps[wave][lane * 32];
      for (int j = 0; j < 32; ++j) {
        const float p = __expf(srow[j] - mx);
        sum += p;
        prow[j] = (_Float16)p;
      }
      row_m[rblk] = mx;
      row_l[rblk] = row_l[rblk] * alpha + sum;
      row_a[rblk] = alpha;
    }
    __syncthreads();

    // rescale running O (co-executes with WMMA), then O += P @ V
    float al[8];
    for (int e = 0; e < 8; ++e) al[e] = row_a[r0 + hi * 8 + e];
    for (int ct = 0; ct < 16; ++ct)
      for (int e = 0; e < 8; ++e) O[ct][e] *= al[e];

    v16h pf = load_a_frag(&Ps[wave][0], 32, 0, 0, lane);
    for (int ct = 0; ct < 16; ++ct) {
      v16h bf = load_b_frag(Vt, 32, ct * 16, 0, lane);
      O[ct] = wmma_f16(pf, bf, O[ct]);
    }
  }
  __syncthreads();

  // normalize, residual add into h (fp32 + f16 shadow)
  float linv[8];
  for (int e = 0; e < 8; ++e) linv[e] = 1.f / row_l[r0 + hi * 8 + e];
  for (int ct = 0; ct < 16; ++ct) {
    const int d = ct * 16 + nn;
    for (int e = 0; e < 8; ++e) {
      const int rblk = r0 + hi * 8 + e;
      const size_t idx = (qrow0 + rblk) * DD + d;
      const float val = h32[idx] + O[ct][e] * linv[e];
      h32[idx] = val;
      h16[idx] = (_Float16)val;
    }
  }
}

// ---------------------------------------------------------------------------
// Host orchestration
// ---------------------------------------------------------------------------
extern "C" void kernel_launch(void* const* d_in, const int* in_sizes, int n_in,
                              void* d_out, int out_size, void* d_ws,
                              size_t ws_size, hipStream_t stream) {
  (void)in_sizes; (void)n_in; (void)out_size; (void)ws_size;
  const int*   x     = (const int*)d_in[0];
  const float* emb   = (const float*)d_in[1];
  const float* Wq    = (const float*)d_in[2];
  const float* bq    = (const float*)d_in[3];
  const float* Wk    = (const float*)d_in[4];
  const float* bk    = (const float*)d_in[5];
  const float* Wv    = (const float*)d_in[6];
  const float* bv    = (const float*)d_in[7];
  const float* W1    = (const float*)d_in[8];
  const float* b1    = (const float*)d_in[9];
  const float* W2    = (const float*)d_in[10];
  const float* b2    = (const float*)d_in[11];
  const float* gamma = (const float*)d_in[12];
  const float* beta  = (const float*)d_in[13];
  const float* Wf    = (const float*)d_in[14];
  const float* bfin  = (const float*)d_in[15];
  float* out = (float*)d_out;

  char* ws = (char*)d_ws;
  float*    h32  = (float*)ws;                                 // 32 MB
  _Float16* h16  = (_Float16*)(ws + (size_t)33554432);         // 16 MB
  _Float16* q16  = (_Float16*)(ws + (size_t)50331648);         // 16 MB
  _Float16* k16  = (_Float16*)(ws + (size_t)67108864);         // 16 MB
  _Float16* v16  = (_Float16*)(ws + (size_t)83886080);         // 16 MB
  _Float16* u16  = (_Float16*)(ws + (size_t)100663296);        // 64 MB
  _Float16* wc16 = (_Float16*)(ws + (size_t)167772160);        // 512 KB scratch

  const float bn_scale = 1.0f / sqrtf(1.0f + 1e-5f);

  embed_kernel<<<MM, DD, 0, stream>>>(x, emb, h32, h16);

  const dim3 gq(MM / 128, DD / 64);   // 256 x 4
  const dim3 gf1(MM / 128, EE / 64);  // 256 x 16
  const dim3 ga(WW / 32, NWW, BB);    // 16 x 8 x 8
  const int cvD = (DD * DD) / (256 * 4);  // conv blocks for DxD weight
  const int cvE = (EE * DD) / (256 * 4);  // conv blocks for ExD weight

  for (int l = 0; l < LL; ++l) {
    const size_t od2 = (size_t)l * DD * DD, oed = (size_t)l * EE * DD;
    convw_kernel<<<cvD, 256, 0, stream>>>(Wq + od2, wc16);
    gemm_kernel<0><<<gq, 256, 0, stream>>>(h16, wc16, bq + (size_t)l * DD,
                                           DD, DD, q16, nullptr, nullptr,
                                           nullptr, 0.f);
    convw_kernel<<<cvD, 256, 0, stream>>>(Wk + od2, wc16);
    gemm_kernel<0><<<gq, 256, 0, stream>>>(h16, wc16, bk + (size_t)l * DD,
                                           DD, DD, k16, nullptr, nullptr,
                                           nullptr, 0.f);
    convw_kernel<<<cvD, 256, 0, stream>>>(Wv + od2, wc16);
    gemm_kernel<0><<<gq, 256, 0, stream>>>(h16, wc16, bv + (size_t)l * DD,
                                           DD, DD, v16, nullptr, nullptr,
                                           nullptr, 0.f);
    attn_kernel<<<ga, 64, 0, stream>>>(q16, k16, v16, h32, h16);
    convw_kernel<<<cvE, 256, 0, stream>>>(W1 + oed, wc16);
    gemm_kernel<1><<<gf1, 256, 0, stream>>>(h16, wc16, b1 + (size_t)l * EE,
                                            DD, EE, u16, nullptr, nullptr,
                                            nullptr, 0.f);
    convw_kernel<<<cvE, 256, 0, stream>>>(W2 + oed, wc16);
    gemm_kernel<2><<<gq, 256, 0, stream>>>(u16, wc16, b2 + (size_t)l * DD,
                                           EE, DD, h16, h32,
                                           gamma + (size_t)l * DD,
                                           beta + (size_t)l * DD, bn_scale);
  }
  convw_kernel<<<cvD, 256, 0, stream>>>(Wf, wc16);
  gemm_kernel<3><<<gq, 256, 0, stream>>>(h16, wc16, bfin, DD, DD, nullptr, out,
                                         nullptr, nullptr, 0.f);
}
